// GCN_80693845557263
// MI455X (gfx1250) — matrix operand order repaired and checked by reading, not compiled, not run on hardware
//
#include <hip/hip_runtime.h>
#include <cstdint>

typedef __attribute__((ext_vector_type(16))) __bf16 v16bf;
typedef __attribute__((ext_vector_type(8)))  __bf16 v8bf;
typedef __attribute__((ext_vector_type(8)))  float  v8f;

#define GK1 1433
#define GC1 32
#define GC2 7
#define KCHUNK 512
#define SSTRIDE 520   // column stride in bf16 elems: 1040 B (16B-aligned, bank-staggered)

static __device__ __forceinline__ __bf16 f2bf(float f) { return (__bf16)f; }
static __device__ __forceinline__ int imin(int a, int b) { return a < b ? a : b; }
static __device__ __forceinline__ v16bf cat8(v8bf lo, v8bf hi) {
    return __builtin_shufflevector(lo, hi, 0, 1, 2, 3, 4, 5, 6, 7,
                                           8, 9, 10, 11, 12, 13, 14, 15);
}

// ---------------- degree / normalization ----------------
__global__ void k_zero(float* __restrict__ p, int n) {
    int i = blockIdx.x * blockDim.x + threadIdx.x;
    if (i < n) p[i] = 0.0f;
}

__global__ void k_degree(const int* __restrict__ dst, float* __restrict__ deg, int E) {
    int e = blockIdx.x * blockDim.x + threadIdx.x;
    if (e < E) atomicAdd(&deg[dst[e]], 1.0f);
}

__global__ void k_rsqrt(float* __restrict__ deg, int n) {
    int i = blockIdx.x * blockDim.x + threadIdx.x;
    if (i < n) deg[i] = rsqrtf(deg[i] + 1.0f);
}

// ---------------- GEMM1 inner k-steps ----------------
// CLAMP only where k can reach >= GK1; k >= GK1 rows of B are zero in LDS so
// clamped (finite) garbage in A contributes exactly 0.
template <bool CLAMP, int NSTEPS>
static __device__ __forceinline__ void gemm1_chunk(const float* __restrict__ xrow,
                                                   const __bf16* __restrict__ bp0,
                                                   const __bf16* __restrict__ bp1,
                                                   int kbase, int k0l0, int half,
                                                   v8f& acc0, v8f& acc1) {
    for (int s = 0; s < NSTEPS; ++s) {
        const int k0l = k0l0 + s * 32;
        v16bf a;
#pragma unroll
        for (int v = 0; v < 8; ++v) {
            int grp = v >> 2, vv = v & 3;
            int k  = kbase + k0l + half * 16 + 2 * v;
            int ka = CLAMP ? imin(k, GK1 - 1) : k;
            int kb = CLAMP ? imin(k + 1, GK1 - 1) : (k + 1);
            a[2 * v]     = f2bf(xrow[ka]);
            a[2 * v + 1] = f2bf(xrow[kb]);
        }
        // B fragments: 16 consecutive bf16 per lane -> 2x ds_load_b128 each
        v8bf lo0 = *(const v8bf*)(bp0 + k0l);
        v8bf hi0 = *(const v8bf*)(bp0 + k0l + 8);
        v8bf lo1 = *(const v8bf*)(bp1 + k0l);
        v8bf hi1 = *(const v8bf*)(bp1 + k0l + 8);
        acc0 = __builtin_amdgcn_wmma_f32_16x16x32_bf16(false, a, false, cat8(lo0, hi0),
                                                       (short)0, acc0, false, false);
        acc1 = __builtin_amdgcn_wmma_f32_16x16x32_bf16(false, a, false, cat8(lo1, hi1),
                                                       (short)0, acc1, false, false);
    }
}

// ---------------- GEMM1: h1 = x @ W1  (bf16 WMMA, f32 accum) ----------------
__global__ __launch_bounds__(256) void k_gemm1(const float* __restrict__ x,
                                               const float* __restrict__ W1,
                                               float* __restrict__ h1, int M) {
    __shared__ __align__(32) __bf16 sW[GC1 * SSTRIDE];  // ~32.5 KB, column-major [n][k]
    const int tid  = threadIdx.x;
    const int wave = tid >> 5;
    const int lane = tid & 31;
    const int half = lane >> 4;
    const int hl   = lane & 15;
    const int row0 = blockIdx.x * 128 + wave * 16;
    const int m    = row0 + hl;
    const int mc   = imin(m, M - 1);            // clamped row: rows >= M never stored
    const float* xrow = x + (long long)mc * GK1;
    const __bf16* bp0 = sW + (size_t)hl * SSTRIDE + half * 16;         // col hl
    const __bf16* bp1 = sW + (size_t)(hl + 16) * SSTRIDE + half * 16;  // col hl+16

    v8f acc0 = {0.f, 0.f, 0.f, 0.f, 0.f, 0.f, 0.f, 0.f};
    v8f acc1 = {0.f, 0.f, 0.f, 0.f, 0.f, 0.f, 0.f, 0.f};

    for (int ch = 0; ch < 3; ++ch) {            // 3 * 512 = 1536 >= 1433
        const int kbase = ch * KCHUNK;
        // stage W1 chunk into LDS column-major as bf16, zero-padding k >= GK1
        for (int idx = tid; idx < KCHUNK * GC1; idx += 256) {
            int klocal = idx >> 5;
            int n      = idx & 31;
            int kk     = kbase + klocal;
            float w = (kk < GK1) ? W1[kk * GC1 + n] : 0.0f;
            sW[n * SSTRIDE + klocal] = f2bf(w);
        }
        __syncthreads();
        if (ch < 2) {
            gemm1_chunk<false, 16>(xrow, bp0, bp1, kbase, 0, half, acc0, acc1);
        } else {
            // chunk 2: k in [1024,1536). Steps 0..11 stay < 1408 <= GK1-1: no clamps.
            gemm1_chunk<false, 12>(xrow, bp0, bp1, kbase, 0,   half, acc0, acc1);
            gemm1_chunk<true,  4>(xrow, bp0, bp1, kbase, 384, half, acc0, acc1);
        }
        __syncthreads();
    }

    if (row0 + 16 <= M) {                        // uniform fast path (all but last block)
#pragma unroll
        for (int r = 0; r < 8; ++r) {
            int rr = row0 + half * 8 + r;
            h1[(long long)rr * GC1 + hl]      = acc0[r];
            h1[(long long)rr * GC1 + 16 + hl] = acc1[r];
        }
    } else {
#pragma unroll
        for (int r = 0; r < 8; ++r) {
            int rr = row0 + half * 8 + r;
            if (rr < M) {
                h1[(long long)rr * GC1 + hl]      = acc0[r];
                h1[(long long)rr * GC1 + 16 + hl] = acc1[r];
            }
        }
    }
}

// ---------------- layer-1 self-loop init + bias ----------------
__global__ void k_init1(const float* __restrict__ h1, const float* __restrict__ dinv,
                        const float* __restrict__ b1, float* __restrict__ agg1, int n) {
    int i = blockIdx.x * blockDim.x + threadIdx.x;
    if (i < n) {
        int node = i >> 5;
        int c    = i & 31;
        float di = dinv[node];
        agg1[i] = h1[i] * di * di + b1[c];
    }
}

// ---------------- layer-1 edge scatter (8 threads / edge) ----------------
__global__ void k_scatter1(const int* __restrict__ src, const int* __restrict__ dst,
                           const float* __restrict__ dinv, const float* __restrict__ h1,
                           float* __restrict__ agg1, int E) {
    int gid = blockIdx.x * blockDim.x + threadIdx.x;
    int e = gid >> 3;
    if (e >= E) return;
    int s = src[e], d = dst[e];
    float nrm = dinv[s] * dinv[d];
    int c0 = (gid & 7) * 4;
    const float4 hv = *reinterpret_cast<const float4*>(h1 + (long long)s * GC1 + c0);
    float* ap = agg1 + (long long)d * GC1 + c0;
    atomicAdd(ap + 0, hv.x * nrm);
    atomicAdd(ap + 1, hv.y * nrm);
    atomicAdd(ap + 2, hv.z * nrm);
    atomicAdd(ap + 3, hv.w * nrm);
}

// ---------------- GEMM2: h2 = relu(agg1) @ W2 (one WMMA per 16 rows) ----------------
__global__ __launch_bounds__(256) void k_gemm2(const float* __restrict__ agg1,
                                               const float* __restrict__ W2,
                                               float* __restrict__ h2, int M) {
    const int tid  = threadIdx.x;
    const int wave = tid >> 5;
    const int lane = tid & 31;
    const int half = lane >> 4;
    const int hl   = lane & 15;
    const int row0 = blockIdx.x * 128 + wave * 16;
    const int m    = row0 + hl;
    const int mc   = imin(m, M - 1);            // clamped row, branchless
    const float* arow = agg1 + (long long)mc * GC1;  // 128B-aligned row

    // A: two runs of 8 consecutive floats -> 4x float4 loads, fused ReLU
    const float4 a0 = *(const float4*)(arow + half * 8);
    const float4 a1 = *(const float4*)(arow + half * 8 + 4);
    const float4 a2 = *(const float4*)(arow + 16 + half * 8);
    const float4 a3 = *(const float4*)(arow + 16 + half * 8 + 4);
    v16bf a;
    a[0]  = f2bf(fmaxf(a0.x, 0.f)); a[1]  = f2bf(fmaxf(a0.y, 0.f));
    a[2]  = f2bf(fmaxf(a0.z, 0.f)); a[3]  = f2bf(fmaxf(a0.w, 0.f));
    a[4]  = f2bf(fmaxf(a1.x, 0.f)); a[5]  = f2bf(fmaxf(a1.y, 0.f));
    a[6]  = f2bf(fmaxf(a1.z, 0.f)); a[7]  = f2bf(fmaxf(a1.w, 0.f));
    a[8]  = f2bf(fmaxf(a2.x, 0.f)); a[9]  = f2bf(fmaxf(a2.y, 0.f));
    a[10] = f2bf(fmaxf(a2.z, 0.f)); a[11] = f2bf(fmaxf(a2.w, 0.f));
    a[12] = f2bf(fmaxf(a3.x, 0.f)); a[13] = f2bf(fmaxf(a3.y, 0.f));
    a[14] = f2bf(fmaxf(a3.z, 0.f)); a[15] = f2bf(fmaxf(a3.w, 0.f));

    // B: 32x16; cols 7..15 hold clamped garbage (finite) — D cols 7..15 never stored,
    // and WMMA columns are independent, so no zeroing is needed.
    v16bf b;
#pragma unroll
    for (int v = 0; v < 8; ++v) {
        int kb = (v >> 2) * 16 + half * 8 + 2 * (v & 3);
        b[2 * v]     = f2bf(W2[imin(kb * GC2 + hl,       32 * GC2 - 1)]);
        b[2 * v + 1] = f2bf(W2[imin((kb + 1) * GC2 + hl, 32 * GC2 - 1)]);
    }
    v8f acc = {0.f, 0.f, 0.f, 0.f, 0.f, 0.f, 0.f, 0.f};
    acc = __builtin_amdgcn_wmma_f32_16x16x32_bf16(false, a, false, b,
                                                  (short)0, acc, false, false);
    if (hl < GC2) {
        if (row0 + 16 <= M) {
#pragma unroll
            for (int r = 0; r < 8; ++r) {
                int rr = row0 + half * 8 + r;
                h2[(long long)rr * 8 + hl] = acc[r];   // stride-8 padded
            }
        } else {
#pragma unroll
            for (int r = 0; r < 8; ++r) {
                int rr = row0 + half * 8 + r;
                if (rr < M) h2[(long long)rr * 8 + hl] = acc[r];
            }
        }
    }
}

// ---------------- layer-2 self-loop init + bias (fully writes d_out) ----------------
__global__ void k_init2(const float* __restrict__ h2, const float* __restrict__ dinv,
                        const float* __restrict__ b2, float* __restrict__ out, int n) {
    int i = blockIdx.x * blockDim.x + threadIdx.x;
    if (i < n) {
        int node = i / GC2;
        int c    = i - node * GC2;
        float di = dinv[node];
        out[i] = h2[(long long)node * 8 + c] * di * di + b2[c];
    }
}

// ---------------- layer-2 edge scatter (1 thread / edge, 7 atomics) ----------------
__global__ void k_scatter2(const int* __restrict__ src, const int* __restrict__ dst,
                           const float* __restrict__ dinv, const float* __restrict__ h2,
                           float* __restrict__ out, int E) {
    int e = blockIdx.x * blockDim.x + threadIdx.x;
    if (e >= E) return;
    int s = src[e], d = dst[e];
    float nrm = dinv[s] * dinv[d];
    const float4 lo = *reinterpret_cast<const float4*>(h2 + (long long)s * 8);
    const float4 hi = *reinterpret_cast<const float4*>(h2 + (long long)s * 8 + 4);
    float* op = out + (long long)d * GC2;
    atomicAdd(op + 0, lo.x * nrm);
    atomicAdd(op + 1, lo.y * nrm);
    atomicAdd(op + 2, lo.z * nrm);
    atomicAdd(op + 3, lo.w * nrm);
    atomicAdd(op + 4, hi.x * nrm);
    atomicAdd(op + 5, hi.y * nrm);
    atomicAdd(op + 6, hi.z * nrm);
}

extern "C" void kernel_launch(void* const* d_in, const int* in_sizes, int n_in,
                              void* d_out, int out_size, void* d_ws, size_t ws_size,
                              hipStream_t stream) {
    const float* x  = (const float*)d_in[0];
    const int*   ei = (const int*)d_in[1];
    const float* W1 = (const float*)d_in[2];
    const float* b1 = (const float*)d_in[3];
    const float* W2 = (const float*)d_in[4];
    const float* b2 = (const float*)d_in[5];

    const int N = in_sizes[0] / GK1;   // 100000
    const int E = in_sizes[1] / 2;     // 3200000
    const int* src = ei;
    const int* dst = ei + E;

    float* ws   = (float*)d_ws;
    float* dinv = ws;                                       // N floats
    float* h1   = ws + (((size_t)N + 255) & ~(size_t)255);  // N*32
    float* agg1 = h1 + (size_t)N * GC1;                     // N*32
    float* h2   = agg1 + (size_t)N * GC1;                   // N*8 (stride-8 padded)
    float* out  = (float*)d_out;                            // N*7

    const int T = 256;
    k_zero    <<<(N + T - 1) / T, T, 0, stream>>>(dinv, N);
    k_degree  <<<(E + T - 1) / T, T, 0, stream>>>(dst, dinv, E);
    k_rsqrt   <<<(N + T - 1) / T, T, 0, stream>>>(dinv, N);
    k_gemm1   <<<(N + 127) / 128, T, 0, stream>>>(x, W1, h1, N);
    k_init1   <<<((size_t)N * GC1 + T - 1) / T, T, 0, stream>>>(h1, dinv, b1, agg1, N * GC1);
    k_scatter1<<<((size_t)E * 8 + T - 1) / T, T, 0, stream>>>(src, dst, dinv, h1, agg1, E);
    k_gemm2   <<<(N + 127) / 128, T, 0, stream>>>(agg1, W2, h2, N);
    k_init2   <<<((size_t)N * GC2 + T - 1) / T, T, 0, stream>>>(h2, dinv, b2, out, N * GC2);
    k_scatter2<<<(E + T - 1) / T, T, 0, stream>>>(src, dst, dinv, h2, out, E);
}
